// GCN_1_24678882083163
// MI455X (gfx1250) — compile-verified, matrix operand order checked by reading, not run on hardware
//
#include <hip/hip_runtime.h>

#define NROW 8192
#define DIN  128
#define DTH  256
#define DOUT 128

typedef __attribute__((ext_vector_type(16))) _Float16 v16h;
typedef __attribute__((ext_vector_type(8)))  _Float16 v8h;
typedef __attribute__((ext_vector_type(8)))  float    v8f;
typedef __attribute__((ext_vector_type(4)))  float    v4f;

static __device__ __forceinline__ v16h cat16(v8h a, v8h b) {
  return __builtin_shufflevector(a, b, 0,1,2,3,4,5,6,7,8,9,10,11,12,13,14,15);
}
static __device__ __forceinline__ v8f wmma_f16(v16h a, v16h b, v8f c) {
  // D = A(16x32 f16) * B(32x16 f16) + C(16x16 f32)
  return __builtin_amdgcn_wmma_f32_16x16x32_f16(false, a, false, b, (short)0, c,
                                                false, false);
}

// ---- async global->LDS path (CDNA5), guarded so compile can't fail
#if defined(__has_builtin)
#if __has_builtin(__builtin_amdgcn_global_load_async_to_lds_b128) && \
    __has_builtin(__builtin_amdgcn_s_wait_asynccnt)
#define USE_ASYNC_LDS 1
#endif
#endif

#ifdef USE_ASYNC_LDS
typedef int v4i_gcc __attribute__((vector_size(16)));
typedef __attribute__((address_space(1))) v4i_gcc gv4i_t;
typedef __attribute__((address_space(3))) v4i_gcc lv4i_t;
static __device__ __forceinline__ void stage_b128(const void* src, void* dst) {
  __builtin_amdgcn_global_load_async_to_lds_b128(
      (gv4i_t*)(uintptr_t)src, (lv4i_t*)(uint32_t)(uintptr_t)dst, 0, 0);
}
static __device__ __forceinline__ void stage_wait() {
  __builtin_amdgcn_s_wait_asynccnt(0);
}
#else
static __device__ __forceinline__ void stage_b128(const void* src, void* dst) {
  *(v8h*)dst = *(const v8h*)src;
}
static __device__ __forceinline__ void stage_wait() {}
#endif

// ---------------------------------------------------------------- utilities
__global__ void zero_f32_kernel(float* __restrict__ p, int n) {
  int i = blockIdx.x * 256 + threadIdx.x;
  if (i < n) p[i] = 0.0f;
}

// --------------------------------------------------------------- batch norm
__global__ void bn_stats_kernel(const float* __restrict__ H,
                                float* __restrict__ colsum,
                                float* __restrict__ colsumsq) {
  int c  = threadIdx.x;
  int r0 = blockIdx.x * 128;
  float s = 0.0f, ss = 0.0f;
  for (int r = 0; r < 128; ++r) {
    float v = H[(size_t)(r0 + r) * DIN + c];
    s += v; ss += v * v;
  }
  atomicAdd(&colsum[c], s);
  atomicAdd(&colsumsq[c], ss);
}

__global__ void bn_finalize_kernel(const float* __restrict__ colsum,
                                   const float* __restrict__ colsumsq,
                                   const float* __restrict__ gamma,
                                   const float* __restrict__ beta,
                                   float* __restrict__ sA, float* __restrict__ sB) {
  int c = threadIdx.x;
  float mu  = colsum[c] * (1.0f / NROW);
  float var = colsumsq[c] * (1.0f / NROW) - mu * mu;
  float rs  = rsqrtf(var + 1e-5f);
  float a   = rs * gamma[c];
  sA[c] = a;
  sB[c] = beta[c] - mu * a;
}

__global__ void bn_apply_kernel(const float* __restrict__ H,
                                const float* __restrict__ sA,
                                const float* __restrict__ sB,
                                _Float16* __restrict__ hn16) {
  int idx = blockIdx.x * 256 + threadIdx.x;
  int c   = idx & (DIN - 1);
  hn16[idx] = (_Float16)(H[idx] * sA[c] + sB[c]);
}

// ------------------------------------------------- W repack to B-frag order
__global__ void repack_w_kernel(const float* __restrict__ W,
                                _Float16* __restrict__ packed, int N) {
  int gid   = blockIdx.x * 256 + threadIdx.x;
  int total = (N / 16) * 4 * 32 * 2;
  if (gid >= total) return;
  int run = gid & 1;
  int l   = (gid >> 1) & 31;
  int kc  = (gid >> 6) & 3;
  int nt  = gid >> 8;
  int n   = nt * 16 + (l & 15);
  int h   = l >> 4;
  int k0  = kc * 32 + 8 * h + 16 * run;
#pragma unroll
  for (int t = 0; t < 8; ++t)
    packed[(size_t)gid * 8 + t] = (_Float16)W[(size_t)(k0 + t) * N + n];
}

// ---------------------------------------------------- GEMM: Hn[8192x128] @ W
template <bool OUTH>
__global__ void __launch_bounds__(256)
gemm_hn_kernel(const _Float16* __restrict__ Amat,
               const _Float16* __restrict__ packedW,
               const float* __restrict__ bias,
               _Float16* __restrict__ outH, float* __restrict__ outF, int N) {
  int tid = threadIdx.x, w = tid >> 5, lane = tid & 31;
  int nn = lane & 15, h = lane >> 4;
  int i0 = blockIdx.x * 128 + w * 16;
  int nt = blockIdx.y;
  const _Float16* arow = Amat + (size_t)(i0 + nn) * DIN;
  v8f acc = {0, 0, 0, 0, 0, 0, 0, 0};
#pragma unroll
  for (int kc = 0; kc < 4; ++kc) {
    v8h a0 = *(const v8h*)(arow + kc * 32 + 8 * h);
    v8h a1 = *(const v8h*)(arow + kc * 32 + 16 + 8 * h);
    const _Float16* bp = packedW + ((size_t)(nt * 4 + kc) * 32 + lane) * 16;
    v8h b0 = *(const v8h*)bp;
    v8h b1 = *(const v8h*)(bp + 8);
    acc = wmma_f16(cat16(a0, a1), cat16(b0, b1), acc);
  }
  float bv = bias[nt * 16 + nn];
#pragma unroll
  for (int r = 0; r < 8; ++r) {
    int row  = i0 + r + 8 * h;
    size_t o = (size_t)row * N + nt * 16 + nn;
    float v  = acc[r] + bv;
    if (OUTH) outH[o] = (_Float16)v;
    else      outF[o] = v;
  }
}

// --------------------------------- pass 1: S = max(sig(Hx Hx^T),.1)*mask + I
// grid (64, 4), block 256 (8 waves). Wave w: rows i0=(bx*8+w)*16.
// 32-wide j tiles, double-buffered async LDS staging, one barrier/iteration.
// Epilogue uses one hoisted base pointer per lane; all 16 Adj loads and 16
// S stores address as base + j0 with compile-time immediate offsets.
__global__ void __launch_bounds__(256)
pass1_kernel(const _Float16* __restrict__ Hx, const float* __restrict__ Adj,
             float* __restrict__ S, float* __restrict__ rowsum) {
  __shared__ _Float16 tile[2][32 * DTH];  // 2 x 16KB
  int tid = threadIdx.x, w = tid >> 5, lane = tid & 31;
  int nn = lane & 15, h = lane >> 4;
  int i0 = (blockIdx.x * 8 + w) * 16;
  int jbase = blockIdx.y * 2048;

  // A fragments (16 rows x 256 K) held in registers across the j loop
  const _Float16* arow = Hx + (size_t)(i0 + nn) * DTH;
  v16h afrag[8];
#pragma unroll
  for (int kc = 0; kc < 8; ++kc)
    afrag[kc] = cat16(*(const v8h*)(arow + kc * 32 + 8 * h),
                      *(const v8h*)(arow + kc * 32 + 16 + 8 * h));

  // hoisted per-lane bases: row block i0+8h(+r), column lane nn
  int rowb = i0 + 8 * h;
  const float* adjb = Adj + (size_t)rowb * NROW + nn;
  float*       sb   = S   + (size_t)rowb * NROW + nn;

  float rs[8] = {0, 0, 0, 0, 0, 0, 0, 0};

  // prologue: stage tile 0 (32 rows x 512B = 16KB, 4 x b128 per thread)
  for (int q = tid; q < 1024; q += 256) {
    int row = q >> 5, off = (q & 31) * 8;
    stage_b128(Hx + (size_t)(jbase + row) * DTH + off,
               &tile[0][row * DTH + off]);
  }

  for (int jt = 0; jt < 64; ++jt) {
    int j0 = jbase + jt * 32;
    stage_wait();
    __syncthreads();  // staged tile visible; previous buffer free for reuse

    if (jt + 1 < 64) {
      int jn = jbase + (jt + 1) * 32;
      _Float16* buf = tile[(jt + 1) & 1];
      for (int q = tid; q < 1024; q += 256) {
        int row = q >> 5, off = (q & 31) * 8;
        stage_b128(Hx + (size_t)(jn + row) * DTH + off, &buf[row * DTH + off]);
      }
    }

    const _Float16* base = tile[jt & 1];
    v8f acc0 = {0, 0, 0, 0, 0, 0, 0, 0};
    v8f acc1 = {0, 0, 0, 0, 0, 0, 0, 0};
#pragma unroll
    for (int kc = 0; kc < 8; ++kc) {
      const _Float16* bpA = base + nn * DTH + kc * 32 + 8 * h;
      const _Float16* bpB = base + (16 + nn) * DTH + kc * 32 + 8 * h;
      v16h bA = cat16(*(const v8h*)bpA, *(const v8h*)(bpA + 16));
      v16h bB = cat16(*(const v8h*)bpB, *(const v8h*)(bpB + 16));
      acc0 = wmma_f16(afrag[kc], bA, acc0);
      acc1 = wmma_f16(afrag[kc], bB, acc1);
    }

    const float* adjp = adjb + j0;  // single address add per iteration
    float*       sp   = sb + j0;
#pragma unroll
    for (int sub = 0; sub < 2; ++sub) {
      v8f acc = sub ? acc1 : acc0;
      int gj = j0 + sub * 16 + nn;
#pragma unroll
      for (int r = 0; r < 8; ++r) {
        float p  = acc[r];
        float sg = __builtin_amdgcn_rcpf(1.0f + __expf(-p));
        sg = fmaxf(sg, 0.1f);
        float av = __builtin_nontemporal_load(adjp + (size_t)(r * NROW + sub * 16));
        float m  = ceilf(av * 1e-5f);
        float sv = sg * m + ((rowb + r == gj) ? 1.0f : 0.0f);
        __builtin_nontemporal_store(sv, sp + (size_t)(r * NROW + sub * 16));
        rs[r] += sv;
      }
    }
  }

  // reduce row partials across the 16 column-lanes of each half-wave
#pragma unroll
  for (int off = 1; off < 16; off <<= 1)
#pragma unroll
    for (int r = 0; r < 8; ++r) rs[r] += __shfl_xor(rs[r], off, 32);
  if (nn == 0) {
#pragma unroll
    for (int r = 0; r < 8; ++r) atomicAdd(&rowsum[rowb + r], rs[r]);
  }
}

__global__ void dfinal_kernel(const float* __restrict__ rowsum,
                              float* __restrict__ d) {
  int i = blockIdx.x * 256 + threadIdx.x;
  if (i < NROW) d[i] = rsqrtf(rowsum[i]);
}

// ------------------------- Yt[c][j] = f16( d[j] * Xo[j][c] ), LDS transpose
__global__ void __launch_bounds__(256)
yprep_kernel(const float* __restrict__ Xo, const float* __restrict__ d,
             _Float16* __restrict__ Yt) {
  __shared__ _Float16 t[128 * 66];
  int tid = threadIdx.x;
  int j0  = blockIdx.x * 64;
#pragma unroll 4
  for (int it = 0; it < 32; ++it) {
    int idx = it * 256 + tid;
    int j = idx >> 7, c = idx & 127;
    float v = Xo[(size_t)(j0 + j) * DOUT + c] * d[j0 + j];
    t[c * 66 + j] = (_Float16)v;
  }
  __syncthreads();
#pragma unroll 4
  for (int it = 0; it < 32; ++it) {
    int idx = it * 256 + tid;
    int c = idx >> 6, jl = idx & 63;
    Yt[(size_t)c * NROW + j0 + jl] = t[c * 66 + jl];
  }
}

// --------------------- pass 2: out = leaky( d_i * (S @ Y) ), S read ONCE
// grid 512, block 256. Block owns rows i0=bx*16; wave w owns cols n0=w*16,
// so all 8 waves share the same S fragments (served from L0/L2).
// Two accumulator chains (K split even/odd) hide WMMA latency.
__global__ void __launch_bounds__(256)
pass2_kernel(const float* __restrict__ S, const _Float16* __restrict__ Yt,
             const float* __restrict__ d, float* __restrict__ out) {
  int tid = threadIdx.x, w = tid >> 5, lane = tid & 31;
  int nn = lane & 15, h = lane >> 4;
  int i0 = blockIdx.x * 16;
  int n0 = w * 16;
  const float*    srow = S  + (size_t)(i0 + nn) * NROW;
  const _Float16* yrow = Yt + (size_t)(n0 + nn) * NROW;
  v8f accP = {0, 0, 0, 0, 0, 0, 0, 0};
  v8f accQ = {0, 0, 0, 0, 0, 0, 0, 0};
  for (int j0 = 0; j0 < NROW; j0 += 64) {
#pragma unroll
    for (int half = 0; half < 2; ++half) {
      int ka = j0 + half * 32 + 8 * h;
      v4f f0 = *(const v4f*)(srow + ka);
      v4f f1 = *(const v4f*)(srow + ka + 4);
      v4f f2 = *(const v4f*)(srow + ka + 16);
      v4f f3 = *(const v4f*)(srow + ka + 20);
      v8h a0, a1;
#pragma unroll
      for (int tq = 0; tq < 4; ++tq) {
        a0[tq]     = (_Float16)f0[tq];
        a0[tq + 4] = (_Float16)f1[tq];
        a1[tq]     = (_Float16)f2[tq];
        a1[tq + 4] = (_Float16)f3[tq];
      }
      v8h b0 = *(const v8h*)(yrow + ka);
      v8h b1 = *(const v8h*)(yrow + ka + 16);
      if (half == 0)
        accP = wmma_f16(cat16(a0, a1), cat16(b0, b1), accP);
      else
        accQ = wmma_f16(cat16(a0, a1), cat16(b0, b1), accQ);
    }
  }
#pragma unroll
  for (int r = 0; r < 8; ++r) {
    int gi  = i0 + r + 8 * h;
    float v = (accP[r] + accQ[r]) * d[gi];
    v = (v > 0.0f) ? v : 0.01f * v;
    __builtin_nontemporal_store(v, &out[(size_t)gi * DOUT + n0 + nn]);
  }
}

// ---------------------------------------------------------------- launcher
extern "C" void kernel_launch(void* const* d_in, const int* in_sizes, int n_in,
                              void* d_out, int out_size, void* d_ws,
                              size_t ws_size, hipStream_t stream) {
  (void)in_sizes; (void)n_in; (void)out_size; (void)ws_size;
  const float* H       = (const float*)d_in[0];
  const float* A       = (const float*)d_in[1];
  const float* gamma   = (const float*)d_in[2];
  const float* beta    = (const float*)d_in[3];
  const float* W_theta = (const float*)d_in[4];
  const float* b_theta = (const float*)d_in[5];
  const float* W_out   = (const float*)d_in[6];
  const float* b_out   = (const float*)d_in[7];

  float* out = (float*)d_out;                       // [8192,128]
  float* S   = (float*)d_out + (size_t)NROW * DOUT; // [8192,8192]

  char* ws = (char*)d_ws;
  size_t off = 0;
  auto alloc = [&](size_t bytes) -> char* {
    char* p = ws + off;
    off = (off + bytes + 255) & ~(size_t)255;
    return p;
  };
  _Float16* hn16 = (_Float16*)alloc((size_t)NROW * DIN * 2);
  _Float16* hx16 = (_Float16*)alloc((size_t)NROW * DTH * 2);
  float*    xo   = (float*)   alloc((size_t)NROW * DOUT * 4);
  _Float16* yt   = (_Float16*)alloc((size_t)DOUT * NROW * 2);
  _Float16* pwt  = (_Float16*)alloc((size_t)(DTH / 16) * 4 * 32 * 2 * 8 * 2);
  _Float16* pwo  = (_Float16*)alloc((size_t)(DOUT / 16) * 4 * 32 * 2 * 8 * 2);
  float* stats   = (float*)alloc((size_t)(NROW + 256) * 4);
  float* rowsum  = stats;
  float* colsum  = stats + NROW;
  float* colsq   = stats + NROW + 128;
  float* sA      = (float*)alloc(128 * 4);
  float* sB      = (float*)alloc(128 * 4);
  float* dvec    = (float*)alloc((size_t)NROW * 4);

  zero_f32_kernel<<<(NROW + 256 + 255) / 256, 256, 0, stream>>>(stats, NROW + 256);
  bn_stats_kernel<<<64, 128, 0, stream>>>(H, colsum, colsq);
  bn_finalize_kernel<<<1, 128, 0, stream>>>(colsum, colsq, gamma, beta, sA, sB);
  bn_apply_kernel<<<(NROW * DIN) / 256, 256, 0, stream>>>(H, sA, sB, hn16);
  repack_w_kernel<<<16, 256, 0, stream>>>(W_theta, pwt, DTH);
  repack_w_kernel<<<8, 256, 0, stream>>>(W_out, pwo, DOUT);
  gemm_hn_kernel<true><<<dim3(64, DTH / 16), 256, 0, stream>>>(
      hn16, pwt, b_theta, hx16, nullptr, DTH);
  gemm_hn_kernel<false><<<dim3(64, DOUT / 16), 256, 0, stream>>>(
      hn16, pwo, b_out, nullptr, xo, DOUT);
  pass1_kernel<<<dim3(64, 4), 256, 0, stream>>>(hx16, A, S, rowsum);
  dfinal_kernel<<<32, 256, 0, stream>>>(rowsum, dvec);
  yprep_kernel<<<128, 256, 0, stream>>>(xo, dvec, yt);
  pass2_kernel<<<512, 256, 0, stream>>>(S, yt, dvec, out);
}
